// TemporalGNN_10522669875753
// MI455X (gfx1250) — matrix operand order repaired and checked by reading, not compiled, not run on hardware
//
#include <hip/hip_runtime.h>
#include <hip/hip_bf16.h>

#define T_   48
#define B_   32
#define N_   96
#define FIN_ 16
#define H_   128
#define C_   500
#define E_   24576           // B*N*DEG
#define M_   147456          // T*B*N
#define G4_  512             // 4*H
#define EPS_ 1e-5f

typedef __bf16 v16bf __attribute__((ext_vector_type(16)));
typedef float  v8f   __attribute__((ext_vector_type(8)));

static __device__ __forceinline__ __bf16 f2bf(float x) { return (__bf16)x; }

static __device__ __forceinline__ void pack8(v16bf& d, int base, float4 a, float4 b) {
  d[base + 0] = f2bf(a.x); d[base + 1] = f2bf(a.y);
  d[base + 2] = f2bf(a.z); d[base + 3] = f2bf(a.w);
  d[base + 4] = f2bf(b.x); d[base + 5] = f2bf(b.y);
  d[base + 6] = f2bf(b.z); d[base + 7] = f2bf(b.w);
}

static __device__ __forceinline__ void load_b16(v16bf& b, const float* p, int ld) {
#pragma unroll
  for (int i = 0; i < 8; ++i) {
    b[i] = f2bf(p[(size_t)i * ld]);
    b[8 + i] = f2bf(p[(size_t)(16 + i) * ld]);
  }
}

// ---------------------------------------------------------------------------
// bf16-WMMA GEMM, NN only: C[M x N] = act(A[M x K] @ B[K x N] + bias + bias2)
// K multiple of 32. Block = 256 thr = 8 waves (2m x 4n), each wave computes a
// 16x32 output strip (two 16x16 WMMA tiles sharing one A fragment) -> block
// tile 32x128.  Fragment layouts per CDNA5 ISA 7.12.2 (wave32):
//   A: lane = M row, elems 0-7 -> k = half*8+i, elems 8-15 -> k = 16+half*8+i
//   B: lane = N column, elems 0-7 -> k = half*8+i, elems 8-15 -> k = 16+half*8+i
//   D: lane -> n = lane&15, VGPR r -> m = r + 8*(lane>>4)
// NG: N-tail guard (wave-uniform scalar branches) — only for FC2 (N=500).
// ---------------------------------------------------------------------------
template <int ACT, bool NG>
__global__ __launch_bounds__(256) void k_gemm(
    const float* __restrict__ A, const float* __restrict__ Bm, float* __restrict__ Cm,
    int Nc, int K, const float* __restrict__ bias, const float* __restrict__ bias2)
{
  const int lane = threadIdx.x & 31;
  const int wave = threadIdx.x >> 5;
  const int m0 = blockIdx.x * 32 + (wave >> 2) * 16;
  const int n0 = blockIdx.y * 128 + (wave & 3) * 32;      // two 16-wide tiles
  const int nlo = lane & 15, nhi = lane >> 4;
  v8f acc0 = {}, acc1 = {};
  const float* arow = A + (size_t)(m0 + nlo) * K + nhi * 8;
  const float* brow = Bm + (size_t)(nhi * 8) * Nc + n0 + nlo;
  for (int kb = 0; kb < K; kb += 32) {
    v16bf a, b0, b1;
    {
      const float4 x0 = ((const float4*)arow)[0];
      const float4 x1 = ((const float4*)arow)[1];
      const float4 x2 = ((const float4*)(arow + 16))[0];
      const float4 x3 = ((const float4*)(arow + 16))[1];
      pack8(a, 0, x0, x1);
      pack8(a, 8, x2, x3);
    }
    if (!NG || n0 + 16 <= Nc) {                           // wave-uniform
      load_b16(b0, brow, Nc);
    } else {
#pragma unroll
      for (int i = 0; i < 16; ++i)                        // uniform per-element guard
        b0[i] = (n0 + nlo < Nc) ? f2bf(brow[(size_t)(i + (i >= 8 ? 8 : 0)) * Nc]) : f2bf(0.f);
    }
    if (!NG || n0 + 32 <= Nc) {
      load_b16(b1, brow + 16, Nc);
    } else {
#pragma unroll
      for (int i = 0; i < 16; ++i)
        b1[i] = (n0 + 16 + nlo < Nc) ? f2bf(brow[16 + (size_t)(i + (i >= 8 ? 8 : 0)) * Nc]) : f2bf(0.f);
    }
    acc0 = __builtin_amdgcn_wmma_f32_16x16x32_bf16(false, a, false, b0, (short)0, acc0,
                                                   false, false);
    acc1 = __builtin_amdgcn_wmma_f32_16x16x32_bf16(false, a, false, b1, (short)0, acc1,
                                                   false, false);
    arow += 32;
    brow += (size_t)32 * Nc;
  }
#pragma unroll
  for (int r = 0; r < 8; ++r) {
    const int m = m0 + r + 8 * nhi;
#pragma unroll
    for (int tile = 0; tile < 2; ++tile) {
      const int n = n0 + tile * 16 + nlo;
      float v = (tile == 0) ? acc0[r] : acc1[r];
      if (!NG || n < Nc) {
        if (bias)  v += bias[n];
        if (bias2) v += bias2[n];
        if (ACT == 1) v = fmaxf(v, 0.f);
        if (ACT == 2) v = tanhf(v);
        Cm[(size_t)m * Nc + n] = v;
      }
    }
  }
}

// ---------------------------------------------------------------------------
// Layout prep: transpose [R][Ccols] -> [Ccols][R]; zero-pad layer-0 operands.
// ---------------------------------------------------------------------------
__global__ void k_transpose(const float* __restrict__ in, float* __restrict__ outp,
                            int R, int Ccols) {
  int i = blockIdx.x * 256 + threadIdx.x;
  if (i >= R * Ccols) return;
  int r = i / Ccols, c = i - r * Ccols;
  outp[(size_t)c * R + r] = in[i];
}

__global__ void k_padx(const float* __restrict__ x, float* __restrict__ xp) {
  int i = blockIdx.x * 256 + threadIdx.x;            // M_*32
  if (i >= M_ * 32) return;
  int m = i >> 5, k = i & 31;
  xp[i] = (k < FIN_) ? x[(size_t)m * FIN_ + k] : 0.f;
}

__global__ void k_padw0(const float* __restrict__ w, float* __restrict__ wp) {
  int i = blockIdx.x * 256 + threadIdx.x;            // 32*H_
  if (i >= 32 * H_) return;
  wp[i] = ((i >> 7) < FIN_) ? w[i] : 0.f;            // row-major [k][128]
}

// ---------------------------------------------------------------------------
// GCN support kernels
// ---------------------------------------------------------------------------
__global__ void k_fill1(float* p, int n) {
  int i = blockIdx.x * 256 + threadIdx.x;
  if (i < n) p[i] = 1.f;
}

__global__ void k_count(const int* __restrict__ ei, float* __restrict__ deg) {
  int i = blockIdx.x * 256 + threadIdx.x;
  if (i >= T_ * E_) return;
  int t = i / E_, e = i - t * E_;
  int dst = ei[(size_t)t * 2 * E_ + E_ + e] + t * (B_ * N_);
  atomicAdd(&deg[dst], 1.f);
}

__global__ void k_dinv(const float* __restrict__ deg, float* __restrict__ dinv, int n) {
  int i = blockIdx.x * 256 + threadIdx.x;
  if (i < n) dinv[i] = rsqrtf(deg[i]);
}

// z = hw * self_norm (self_norm = 1/deg = dinv^2) — zero-bases the scatter target
__global__ void k_self(const float* __restrict__ hw, const float* __restrict__ dinv,
                       float* __restrict__ z) {
  size_t i = (size_t)blockIdx.x * 256 + threadIdx.x;
  if (i >= (size_t)M_ * H_) return;
  int node = (int)(i >> 7);
  float d = dinv[node];
  z[i] = hw[i] * d * d;
}

// z[dst] += hw[src] * dinv[src]*dinv[dst]; one thread per (edge, float4 chunk).
// z (75 MB) resides in the 192 MB L2 -> atomics stay off HBM.
__global__ void k_scatter(const int* __restrict__ ei, const float* __restrict__ dinv,
                          const float* __restrict__ hw, float* __restrict__ z) {
  size_t tid = (size_t)blockIdx.x * 256 + threadIdx.x;
  if (tid >= (size_t)T_ * E_ * (H_ / 4)) return;
  int q = (int)(tid & 31);
  size_t e = tid >> 5;
  int t = (int)(e / E_), el = (int)(e - (size_t)t * E_);
  const int* base = ei + (size_t)t * 2 * E_;
  const int off = t * (B_ * N_);
  const int src = base[el] + off;
  const int dst = base[E_ + el] + off;
  const float nrm = dinv[src] * dinv[dst];
  __builtin_prefetch(hw + (size_t)src * H_, 0, 1);   // global_prefetch_b8
  const float4 v = ((const float4*)(hw + (size_t)src * H_))[q];
  float* zp = z + (size_t)dst * H_ + q * 4;
  atomicAdd(zp + 0, v.x * nrm);
  atomicAdd(zp + 1, v.y * nrm);
  atomicAdd(zp + 2, v.z * nrm);
  atomicAdd(zp + 3, v.w * nrm);
}

// h = relu(BN(z + gcn_b)) [+ h if res]
__global__ void k_bn(const float* __restrict__ z, float* __restrict__ h,
                     const float* __restrict__ gb, const float* __restrict__ gam,
                     const float* __restrict__ bet, const float* __restrict__ mu,
                     const float* __restrict__ var, int res) {
  size_t i = (size_t)blockIdx.x * 256 + threadIdx.x;
  if (i >= (size_t)M_ * H_) return;
  int hh = (int)(i & (H_ - 1));
  float sc = gam[hh] * rsqrtf(var[hh] + EPS_);
  float v = (z[i] + gb[hh] - mu[hh]) * sc + bet[hh];
  v = fmaxf(v, 0.f);
  h[i] = res ? (v + h[i]) : v;
}

__global__ void k_pool(const float* __restrict__ h, float* __restrict__ emb) {
  int i = blockIdx.x * 256 + threadIdx.x;
  if (i >= T_ * B_ * H_) return;
  int tb = i / H_, hh = i - tb * H_;
  const float* base = h + (size_t)tb * N_ * H_ + hh;
  float s = 0.f;
  for (int n = 0; n < N_; ++n) s += base[(size_t)n * H_];
  emb[i] = s * (1.f / N_);
}

// ---------------------------------------------------------------------------
// BiLSTM recurrent scan. Grid = 2 blocks (fwd, bwd), 512 threads = 16 waves.
// Each wave owns a (16 batch x 16 hidden) cell tile -> computes all 4 gate
// tiles with WMMA (h in LDS as A via ds_load_b128, whh^T pre-transposed so
// each B-fragment k row is 64 contiguous bytes over 16 lanes), c persistent in regs.
// ---------------------------------------------------------------------------
__global__ __launch_bounds__(512) void k_lstm_bidir(
    const float* __restrict__ preF, const float* __restrict__ preB,
    const float* __restrict__ whhTF, const float* __restrict__ whhTB,  // [H][4H]
    float* __restrict__ out, int out_ld)
{
  __shared__ float hbuf[B_][H_];                 // 16 KB
  const int rev = blockIdx.x;                    // 0 = fwd, 1 = bwd
  const float* pre  = rev ? preB  : preF;
  const float* whhT = rev ? whhTB : whhTF;
  const int ooff = rev ? H_ : 0;
  const int tid = threadIdx.x;
  const int lane = tid & 31;
  const int wave = tid >> 5;                     // 0..15
  const int mhalf = wave >> 3;                   // batch half
  const int nsl = wave & 7;                      // hidden 16-slice
  const int nlo = lane & 15, nhi = lane >> 4;

  for (int i = tid; i < B_ * H_; i += 512) (&hbuf[0][0])[i] = 0.f;
  float c[8];
#pragma unroll
  for (int r = 0; r < 8; ++r) c[r] = 0.f;
  __syncthreads();

  const int m = mhalf * 16 + nlo;
  const int j = nsl * 16 + nlo;

  for (int s = 0; s < T_; ++s) {
    const int t = rev ? (T_ - 1 - s) : s;
    v8f acc[4] = {};
    for (int kb = 0; kb < H_; kb += 32) {
      v16bf a;
      {
        const float* hr = &hbuf[m][kb + nhi * 8];
        const float4 x0 = ((const float4*)hr)[0];
        const float4 x1 = ((const float4*)hr)[1];
        const float4 x2 = ((const float4*)(hr + 16))[0];
        const float4 x3 = ((const float4*)(hr + 16))[1];
        pack8(a, 0, x0, x1);
        pack8(a, 8, x2, x3);
      }
      const float* br = whhT + (size_t)(kb + nhi * 8) * G4_ + nsl * 16 + nlo;
#pragma unroll
      for (int g = 0; g < 4; ++g) {
        v16bf b;
        load_b16(b, br + g * H_, G4_);
        acc[g] = __builtin_amdgcn_wmma_f32_16x16x32_bf16(false, a, false, b, (short)0,
                                                         acc[g], false, false);
      }
    }
    __syncthreads();                              // all hbuf reads complete
#pragma unroll
    for (int r = 0; r < 8; ++r) {
      const int brow = mhalf * 16 + r + 8 * nhi;  // D layout: m = r + 8*half
      const float* p = pre + ((size_t)t * B_ + brow) * G4_;
      const float gi = acc[0][r] + p[0 * H_ + j];
      const float gf = acc[1][r] + p[1 * H_ + j];
      const float gg = acc[2][r] + p[2 * H_ + j];
      const float go = acc[3][r] + p[3 * H_ + j];
      const float si = 1.f / (1.f + __expf(-gi));
      const float sf = 1.f / (1.f + __expf(-gf));
      const float so = 1.f / (1.f + __expf(-go));
      c[r] = sf * c[r] + si * tanhf(gg);
      const float hv = so * tanhf(c[r]);
      hbuf[brow][j] = hv;
      out[((size_t)t * B_ + brow) * out_ld + ooff + j] = hv;
    }
    __syncthreads();                              // h published for next step
  }
}

// ---------------------------------------------------------------------------
// Attention + head
// ---------------------------------------------------------------------------
__global__ void k_attn_score(const float* __restrict__ a1, const float* __restrict__ w2,
                             const float* __restrict__ b2, float* __restrict__ score) {
  int row = blockIdx.x * 8 + (threadIdx.x >> 5);  // 1536 rows, 1 wave each
  int lane = threadIdx.x & 31;
  float s = 0.f;
  for (int h = lane; h < H_; h += 32) s += a1[(size_t)row * H_ + h] * w2[h];
  for (int off = 16; off; off >>= 1) s += __shfl_down(s, off, 32);
  if (lane == 0) score[row] = s + b2[0];
}

__global__ void k_attn_pool(const float* __restrict__ score, const float* __restrict__ out2,
                            float* __restrict__ wsum) {
  __shared__ float w[T_];
  __shared__ float invs;
  const int b = blockIdx.x;
  if (threadIdx.x < T_) w[threadIdx.x] = score[threadIdx.x * B_ + b];
  __syncthreads();
  if (threadIdx.x == 0) {
    float mx = w[0];
    for (int t = 1; t < T_; ++t) mx = fmaxf(mx, w[t]);
    float sm = 0.f;
    for (int t = 0; t < T_; ++t) { w[t] = __expf(w[t] - mx); sm += w[t]; }
    invs = 1.f / sm;
  }
  __syncthreads();
  const float inv = invs;
  for (int h = threadIdx.x; h < 2 * H_; h += blockDim.x) {
    float acc = 0.f;
    for (int t = 0; t < T_; ++t) acc += w[t] * out2[((size_t)t * B_ + b) * (2 * H_) + h];
    wsum[b * (2 * H_) + h] = acc * inv;
  }
}

__global__ void k_logsoftmax(float* __restrict__ y) {
  __shared__ float red[256];
  const int b = blockIdx.x;
  float mx = -1e30f;
  for (int c = threadIdx.x; c < C_; c += 256) mx = fmaxf(mx, y[(size_t)b * C_ + c]);
  red[threadIdx.x] = mx; __syncthreads();
  for (int s = 128; s; s >>= 1) {
    if (threadIdx.x < s) red[threadIdx.x] = fmaxf(red[threadIdx.x], red[threadIdx.x + s]);
    __syncthreads();
  }
  mx = red[0]; __syncthreads();
  float sm = 0.f;
  for (int c = threadIdx.x; c < C_; c += 256) sm += __expf(y[(size_t)b * C_ + c] - mx);
  red[threadIdx.x] = sm; __syncthreads();
  for (int s = 128; s; s >>= 1) {
    if (threadIdx.x < s) red[threadIdx.x] += red[threadIdx.x + s];
    __syncthreads();
  }
  const float lse = mx + logf(red[0]);
  __syncthreads();
  for (int c = threadIdx.x; c < C_; c += 256) y[(size_t)b * C_ + c] -= lse;
}

// ---------------------------------------------------------------------------
extern "C" void kernel_launch(void* const* d_in, const int* in_sizes, int n_in,
                              void* d_out, int out_size, void* d_ws, size_t ws_size,
                              hipStream_t stream) {
  (void)in_sizes; (void)n_in; (void)out_size; (void)ws_size;
  const float* x        = (const float*)d_in[0];
  const int*   eidx     = (const int*)  d_in[1];
  const float* gcn_w0   = (const float*)d_in[2];
  const float* gcn_w12  = (const float*)d_in[3];
  const float* gcn_b    = (const float*)d_in[4];
  const float* bn_gam   = (const float*)d_in[5];
  const float* bn_bet   = (const float*)d_in[6];
  const float* bn_mu    = (const float*)d_in[7];
  const float* bn_var   = (const float*)d_in[8];
  const float* wih0     = (const float*)d_in[9];
  const float* whh0     = (const float*)d_in[10];
  const float* bih0     = (const float*)d_in[11];
  const float* bhh0     = (const float*)d_in[12];
  const float* wih1     = (const float*)d_in[13];
  const float* whh1     = (const float*)d_in[14];
  const float* bih1     = (const float*)d_in[15];
  const float* bhh1     = (const float*)d_in[16];
  const float* attn_w1  = (const float*)d_in[17];
  const float* attn_b1  = (const float*)d_in[18];
  const float* attn_w2  = (const float*)d_in[19];
  const float* attn_b2  = (const float*)d_in[20];
  const float* fc1_w    = (const float*)d_in[21];
  const float* fc1_b    = (const float*)d_in[22];
  const float* fc2_w    = (const float*)d_in[23];
  const float* fc2_b    = (const float*)d_in[24];
  float* out = (float*)d_out;

  size_t off = 0;
  auto carve = [&](size_t bytes) -> float* {
    off = (off + 255) & ~(size_t)255;
    float* r = (float*)((char*)d_ws + off);
    off += bytes;
    return r;
  };
  float* hbuf  = carve((size_t)M_ * H_ * 4);
  float* hw    = carve((size_t)M_ * H_ * 4);
  float* z     = carve((size_t)M_ * H_ * 4);
  float* xpad  = carve((size_t)M_ * 32 * 4);
  float* w0pad = carve((size_t)32 * H_ * 4);
  float* deg   = carve((size_t)M_ * 4);
  float* dinv  = carve((size_t)M_ * 4);
  float* emb   = carve((size_t)T_ * B_ * H_ * 4);
  float* preF  = carve((size_t)T_ * B_ * G4_ * 4);
  float* preB  = carve((size_t)T_ * B_ * G4_ * 4);
  float* out1  = carve((size_t)T_ * B_ * 2 * H_ * 4);
  float* out2  = carve((size_t)T_ * B_ * 2 * H_ * 4);
  float* a1    = carve((size_t)T_ * B_ * H_ * 4);
  float* scr   = carve((size_t)T_ * B_ * 4);
  float* wsum  = carve((size_t)B_ * 2 * H_ * 4);
  float* fc1o  = carve((size_t)B_ * H_ * 4);
  float* wih0T = carve((size_t)2 * H_ * G4_ * 4);        // [H][4H] x2 dirs
  float* wih1T = carve((size_t)2 * 2 * H_ * G4_ * 4);    // [2H][4H] x2 dirs
  float* whh0T = carve((size_t)2 * H_ * G4_ * 4);        // [H][4H] x2 dirs
  float* whh1T = carve((size_t)2 * H_ * G4_ * 4);

  auto gemm = [&](const float* A, const float* Bm, float* Cm, int Mr, int Nc, int K,
                  const float* b1, const float* b2, int act, bool ng) {
    dim3 g(Mr / 32, (Nc + 127) / 128);
    if (ng)             k_gemm<0, true ><<<g, 256, 0, stream>>>(A, Bm, Cm, Nc, K, b1, b2);
    else if (act == 1)  k_gemm<1, false><<<g, 256, 0, stream>>>(A, Bm, Cm, Nc, K, b1, b2);
    else if (act == 2)  k_gemm<2, false><<<g, 256, 0, stream>>>(A, Bm, Cm, Nc, K, b1, b2);
    else                k_gemm<0, false><<<g, 256, 0, stream>>>(A, Bm, Cm, Nc, K, b1, b2);
  };

  // ---- layout prep: padding + weight transposes (cheap, one pass each) ----
  k_padx<<<(M_ * 32 + 255) / 256, 256, 0, stream>>>(x, xpad);
  k_padw0<<<(32 * H_ + 255) / 256, 256, 0, stream>>>(gcn_w0, w0pad);
  for (int d = 0; d < 2; ++d) {
    k_transpose<<<(G4_ * H_ + 255) / 256, 256, 0, stream>>>(
        wih0 + (size_t)d * G4_ * H_, wih0T + (size_t)d * G4_ * H_, G4_, H_);
    k_transpose<<<(G4_ * 2 * H_ + 255) / 256, 256, 0, stream>>>(
        wih1 + (size_t)d * G4_ * 2 * H_, wih1T + (size_t)d * G4_ * 2 * H_, G4_, 2 * H_);
    k_transpose<<<(G4_ * H_ + 255) / 256, 256, 0, stream>>>(
        whh0 + (size_t)d * G4_ * H_, whh0T + (size_t)d * G4_ * H_, G4_, H_);
    k_transpose<<<(G4_ * H_ + 255) / 256, 256, 0, stream>>>(
        whh1 + (size_t)d * G4_ * H_, whh1T + (size_t)d * G4_ * H_, G4_, H_);
  }

  // ---- degree / norms ----
  k_fill1<<<(M_ + 255) / 256, 256, 0, stream>>>(deg, M_);
  k_count<<<(T_ * E_ + 255) / 256, 256, 0, stream>>>(eidx, deg);
  k_dinv<<<(M_ + 255) / 256, 256, 0, stream>>>(deg, dinv, M_);

  // ---- GCN layers ----
  const size_t MH = (size_t)M_ * H_;
  for (int l = 0; l < 3; ++l) {
    const float* A = (l == 0) ? xpad : hbuf;
    const float* W = (l == 0) ? w0pad : (gcn_w12 + (size_t)(l - 1) * H_ * H_);
    const int K = (l == 0) ? 32 : H_;
    gemm(A, W, hw, M_, H_, K, nullptr, nullptr, 0, false);
    k_self<<<(unsigned)(MH / 256), 256, 0, stream>>>(hw, dinv, z);
    k_scatter<<<(unsigned)((size_t)T_ * E_ * 32 / 256), 256, 0, stream>>>(eidx, dinv, hw, z);
    k_bn<<<(unsigned)(MH / 256), 256, 0, stream>>>(z, hbuf, gcn_b + l * H_, bn_gam + l * H_,
                                                   bn_bet + l * H_, bn_mu + l * H_,
                                                   bn_var + l * H_, l > 0 ? 1 : 0);
  }
  k_pool<<<(T_ * B_ * H_ + 255) / 256, 256, 0, stream>>>(hbuf, emb);

  // ---- BiLSTM layer 0 (input H) ----
  gemm(emb, wih0T, preF, T_ * B_, G4_, H_, bih0, bhh0, 0, false);
  gemm(emb, wih0T + (size_t)G4_ * H_, preB, T_ * B_, G4_, H_,
       bih0 + G4_, bhh0 + G4_, 0, false);
  k_lstm_bidir<<<2, 512, 0, stream>>>(preF, preB, whh0T, whh0T + (size_t)G4_ * H_,
                                      out1, 2 * H_);
  // ---- BiLSTM layer 1 (input 2H) ----
  gemm(out1, wih1T, preF, T_ * B_, G4_, 2 * H_, bih1, bhh1, 0, false);
  gemm(out1, wih1T + (size_t)G4_ * 2 * H_, preB, T_ * B_, G4_, 2 * H_,
       bih1 + G4_, bhh1 + G4_, 0, false);
  k_lstm_bidir<<<2, 512, 0, stream>>>(preF, preB, whh1T, whh1T + (size_t)G4_ * H_,
                                      out2, 2 * H_);

  // ---- attention ----
  gemm(out2, attn_w1, a1, T_ * B_, H_, 2 * H_, attn_b1, nullptr, 2 /*tanh*/, false);
  k_attn_score<<<T_ * B_ / 8, 256, 0, stream>>>(a1, attn_w2, attn_b2, scr);
  k_attn_pool<<<B_, 256, 0, stream>>>(scr, out2, wsum);

  // ---- head ----
  gemm(wsum, fc1_w, fc1o, B_, H_, 2 * H_, fc1_b, nullptr, 1 /*relu*/, false);
  gemm(fc1o, fc2_w, out, B_, C_, H_, fc2_b, nullptr, 0, true /*N guard*/);
  k_logsoftmax<<<B_, 256, 0, stream>>>(out);
}
